// SpectralStateLM_18631568130206
// MI455X (gfx1250) — matrix operand order repaired and checked by reading, not compiled, not run on hardware
//
#include <hip/hip_runtime.h>
#include <hip/hip_bf16.h>
#include <math.h>

// ---------------- problem constants (fixed by setup_inputs) ----------------
#define BB 8
#define TT 2048
#define VV 32000
#define SS 3
#define KK 4
#define SK 12            // S*K
#define NN 4096          // masked positions = B*T/4
#define NCH 16           // scan chunks
#define CL 128           // chunk length (NCH*CL == TT)
#define NSEQ 192         // 2 channels * B * S * K
#define NTILES (NN / 16)     // 256
#define VTILES (VV / 16)     // 2000
#define VGROUPS (VTILES / 8) // 250 ; each wave does 8 v-tiles

typedef __attribute__((ext_vector_type(16))) _Float16 v16h;
typedef __attribute__((ext_vector_type(8)))  float    v8f;

__device__ __forceinline__ float sigm(float x) { return 1.0f / (1.0f + __expf(-x)); }

// ---------------- Phase 0: vocab feature matrix (V,32) f16, scaled by 1/|temp| ---
__global__ __launch_bounds__(256) void vfeats_kernel(const float* __restrict__ damp,
                                                     const float* __restrict__ dph,
                                                     const float* __restrict__ temp,
                                                     _Float16* __restrict__ vf) {
  int tid = blockIdx.x * blockDim.x + threadIdx.x;
  if (tid >= VV * SK) return;
  int v = tid / SK, j = tid - v * SK;
  float sc  = 1.0f / (fabsf(temp[0]) + 1e-6f);
  float amp = damp[v * SK + j] * sc;
  float ph  = dph[v * SK + j];
  vf[v * 32 + j]      = (_Float16)(amp * cosf(ph));
  vf[v * 32 + 12 + j] = (_Float16)(amp * sinf(ph));
  if (j < 8) vf[v * 32 + 24 + j] = (_Float16)0.0f;   // zero-pad K 24..31
}

// seq decode: c (amp/phase), b (batch), j=s*4+k
__device__ __forceinline__ void seq_decode(int seq, int& c, int& b, int& j, int& s) {
  c = seq / (BB * SK);
  int rem = seq - c * (BB * SK);
  b = rem / SK;
  j = rem - b * SK;
  s = j >> 2;
}

// ---------------- Phase 1: per-chunk affine reduction (A, Uf, Ub) ---------------
__global__ __launch_bounds__(256) void scan_local_kernel(const float* __restrict__ damp,
                                                         const float* __restrict__ dph,
                                                         const float* __restrict__ dlog,
                                                         const int*   __restrict__ tok,
                                                         float* __restrict__ A,
                                                         float* __restrict__ Uf,
                                                         float* __restrict__ Ub) {
  int gid = blockIdx.x * blockDim.x + threadIdx.x;
  if (gid >= NSEQ * NCH) return;
  int chunk = gid & (NCH - 1), seq = gid >> 4;
  int c, b, j, s; seq_decode(seq, c, b, j, s);
  const float* delta = c ? dph : damp;
  const int*   tb    = tok + b * TT;
  float dec = sigm(dlog[s]);
  int t0 = chunk * CL;

  float Af = 1.0f, Ufv = 0.0f;
  for (int t = t0; t < t0 + CL; ++t) {
    bool msk = ((t & 3) == 1);
    float d = msk ? 1.0f : dec;
    float u = msk ? 0.0f : delta[tb[t] * SK + j];
    Ufv = d * Ufv + u;
    Af *= d;
  }
  float Ubv = 0.0f;
  for (int t = t0 + CL - 1; t >= t0; --t) {
    bool msk = ((t & 3) == 1);
    float d = msk ? 1.0f : dec;
    float u = msk ? 0.0f : delta[tb[t] * SK + j];
    Ubv = d * Ubv + u;
  }
  A[gid] = Af; Uf[gid] = Ufv; Ub[gid] = Ubv;
}

// ---------------- Phase 2: serial prefix over 16 chunks per sequence ------------
__global__ __launch_bounds__(256) void chunk_prefix_kernel(const float* __restrict__ A,
                                                           const float* __restrict__ Uf,
                                                           const float* __restrict__ Ub,
                                                           float* __restrict__ Sf,
                                                           float* __restrict__ Sb) {
  int seq = blockIdx.x * blockDim.x + threadIdx.x;
  if (seq >= NSEQ) return;
  float st = 0.0f;
  for (int ch = 0; ch < NCH; ++ch) {
    int idx = seq * NCH + ch;
    Sf[idx] = st;
    st = A[idx] * st + Uf[idx];
  }
  st = 0.0f;
  for (int ch = NCH - 1; ch >= 0; --ch) {
    int idx = seq * NCH + ch;
    Sb[idx] = st;
    st = A[idx] * st + Ub[idx];
  }
}

// ---------------- Phase 3: re-emit combined states at masked positions ----------
__global__ __launch_bounds__(256) void emit_comb_kernel(const float* __restrict__ damp,
                                                        const float* __restrict__ dph,
                                                        const float* __restrict__ dlog,
                                                        const float* __restrict__ cw,
                                                        const int*   __restrict__ tok,
                                                        const float* __restrict__ Sf,
                                                        const float* __restrict__ Sb,
                                                        float* __restrict__ comb) {
  int gid = blockIdx.x * blockDim.x + threadIdx.x;
  if (gid >= NSEQ * NCH) return;
  int chunk = gid & (NCH - 1), seq = gid >> 4;
  int c, b, j, s; seq_decode(seq, c, b, j, s);
  const float* delta = c ? dph : damp;
  const int*   tb    = tok + b * TT;
  float dec = sigm(dlog[s]);
  float w   = sigm(cw[0]);
  int t0 = chunk * CL;

  // forward exclusive scan; stash F_t at the 32 masked steps of this chunk
  float Fm[CL / 4];
  float carry = Sf[gid];
  int mi = 0;
  for (int t = t0; t < t0 + CL; ++t) {
    bool msk = ((t & 3) == 1);
    if (msk) Fm[mi++] = carry;
    float d = msk ? 1.0f : dec;
    float u = msk ? 0.0f : delta[tb[t] * SK + j];
    carry = d * carry + u;
  }
  // backward exclusive scan; combine and write comb[channel][n][j]
  carry = Sb[gid];
  mi = CL / 4 - 1;
  float* combc = comb + (size_t)c * NN * SK;
  for (int t = t0 + CL - 1; t >= t0; --t) {
    bool msk = ((t & 3) == 1);
    if (msk) {
      int n = (t >> 2) * BB + b;          // masked_idx ordering: t-major then batch
      combc[n * SK + j] = w * Fm[mi--] + (1.0f - w) * carry;
    }
    float d = msk ? 1.0f : dec;
    float u = msk ? 0.0f : delta[tb[t] * SK + j];
    carry = d * carry + u;
  }
}

// ---------------- Phase 4: masked-position feature matrix (N,32) f16 ------------
__global__ __launch_bounds__(256) void feats_kernel(const float* __restrict__ comb,
                                                    _Float16* __restrict__ fe) {
  int tid = blockIdx.x * blockDim.x + threadIdx.x;
  if (tid >= NN * SK) return;
  int n = tid / SK, j = tid - n * SK;
  float amp = comb[n * SK + j];
  float ph  = comb[(size_t)NN * SK + n * SK + j];
  fe[n * 32 + j]      = (_Float16)(amp * cosf(ph));
  fe[n * 32 + 12 + j] = (_Float16)(amp * sinf(ph));
  if (j < 8) fe[n * 32 + 24 + j] = (_Float16)0.0f;
}

// ---------------- Phase 5: WMMA GEMM  out(4096x32000) = feats @ vfeats^T --------
// One wave -> one 16-row N-tile x 8 consecutive 16-col V-tiles (A-frag reused).
// Output is a 524MB pure streaming write: store non-temporal so the logits
// stream does not evict the WMMA operand tiles (vfeats 2MB) from L2.
__global__ __launch_bounds__(256) void gemm_wmma_kernel(const _Float16* __restrict__ fe,
                                                        const _Float16* __restrict__ vf,
                                                        float* __restrict__ out) {
  const int lane = threadIdx.x & 31;
  const int wave = threadIdx.x >> 5;
  const int unit = blockIdx.x * 8 + wave;          // [0, NTILES*VGROUPS)
  const int ntile  = unit / VGROUPS;
  const int vgroup = unit - ntile * VGROUPS;
  const int h = lane >> 4;                         // half-wave select
  const int m = lane & 15;

  // A fragment: row M=m, halves K = [16h .. 16h+15] (contiguous 32B)
  const _Float16* ap = fe + (size_t)(ntile * 16 + m) * 32 + 16 * h;
  v16h a;
#pragma unroll
  for (int i = 0; i < 8; ++i) { a[i] = ap[i]; a[8 + i] = ap[8 + i]; }

#pragma unroll
  for (int vt = 0; vt < 8; ++vt) {
    int col0 = (vgroup * 8 + vt) * 16;
    // B fragment: col N=m, halves K = [16h .. 16h+15] (contiguous 32B)
    const _Float16* bp = vf + (size_t)(col0 + m) * 32 + 16 * h;
    v16h bfrag;
#pragma unroll
    for (int i = 0; i < 16; ++i) bfrag[i] = bp[i];

    v8f acc = {};
    acc = __builtin_amdgcn_wmma_f32_16x16x32_f16(
        /*neg_a=*/false, a, /*neg_b=*/false, bfrag,
        /*c_mod=*/(short)0, acc, /*reuse_a=*/false, /*reuse_b=*/false);

    float* op = out + (size_t)(ntile * 16 + 8 * h) * VV + col0 + m;
#pragma unroll
    for (int r = 0; r < 8; ++r)
      __builtin_nontemporal_store(acc[r], op + (size_t)r * VV);
  }
}

// ---------------- launcher ------------------------------------------------------
extern "C" void kernel_launch(void* const* d_in, const int* in_sizes, int n_in,
                              void* d_out, int out_size, void* d_ws, size_t ws_size,
                              hipStream_t stream) {
  const float* damp = (const float*)d_in[0];   // (V,S,K)
  const float* dph  = (const float*)d_in[1];   // (V,S,K)
  const float* dlog = (const float*)d_in[2];   // (S,)
  const float* cw   = (const float*)d_in[3];   // scalar
  const float* temp = (const float*)d_in[4];   // scalar
  const int*   tok  = (const int*)d_in[5];     // (B,T)
  float* out = (float*)d_out;

  // workspace carve-up (bytes); ~2.77 MB total, all 16B-aligned
  char* base = (char*)d_ws;
  float*    wsA   = (float*)(base + 0);          // 3072 f
  float*    wsUf  = (float*)(base + 12288);      // 3072 f
  float*    wsUb  = (float*)(base + 24576);      // 3072 f
  float*    wsSf  = (float*)(base + 36864);      // 3072 f
  float*    wsSb  = (float*)(base + 49152);      // 3072 f
  float*    wsCmb = (float*)(base + 61440);      // 2*4096*12 f = 393216 B
  _Float16* wsFe  = (_Float16*)(base + 454656);  // 4096*32 h   = 262144 B
  _Float16* wsVf  = (_Float16*)(base + 716800);  // 32000*32 h  = 2048000 B

  vfeats_kernel<<<(VV * SK + 255) / 256, 256, 0, stream>>>(damp, dph, temp, wsVf);
  scan_local_kernel<<<(NSEQ * NCH + 255) / 256, 256, 0, stream>>>(
      damp, dph, dlog, tok, wsA, wsUf, wsUb);
  chunk_prefix_kernel<<<1, 256, 0, stream>>>(wsA, wsUf, wsUb, wsSf, wsSb);
  emit_comb_kernel<<<(NSEQ * NCH + 255) / 256, 256, 0, stream>>>(
      damp, dph, dlog, cw, tok, wsSf, wsSb, wsCmb);
  feats_kernel<<<(NN * SK + 255) / 256, 256, 0, stream>>>(wsCmb, wsFe);
  gemm_wmma_kernel<<<(NTILES * VGROUPS) / 8, 256, 0, stream>>>(wsFe, wsVf, out);
}
